// ModulatedDeformRoIPoolingPack_14516989461143
// MI455X (gfx1250) — matrix operand order repaired and checked by reading, not compiled, not run on hardware
//
#include <hip/hip_runtime.h>
#include <hip/hip_bf16.h>
#include <math.h>

#define SPATIAL_SCALE 0.0625f
#define POOLED 7
#define SAMPLE 4
#define TRANS_STD 0.1f

static constexpr int B_ = 2, C_ = 256, H_ = 100, W_ = 152, N_ = 128, FC = 1024;
static constexpr int PP = POOLED * POOLED;          // 49
static constexpr int IN = PP * C_;                  // 12544

typedef __attribute__((ext_vector_type(16))) _Float16 v16h;
typedef __attribute__((ext_vector_type(8)))  _Float16 v8h;
typedef __attribute__((ext_vector_type(8)))  float    v8f;

// ---------------------------------------------------------------------------
// BCHW -> BHWC transpose (so pooling gathers read 256 contiguous channels)
// ---------------------------------------------------------------------------
__global__ void transpose_bchw_to_bhwc(const float* __restrict__ src,
                                       float* __restrict__ dst) {
  __shared__ float tile[32][33];
  const int by = blockIdx.z;                 // b*H + y
  const int b = by / H_, y = by % H_;
  const int x0 = blockIdx.x * 32;
  const int c0 = blockIdx.y * 32;
  const int tx = threadIdx.x, ty = threadIdx.y;
#pragma unroll
  for (int j = 0; j < 4; ++j) {
    int c = c0 + ty + j * 8;
    int x = x0 + tx;
    float v = 0.f;
    if (x < W_) v = src[(((size_t)b * C_ + c) * H_ + y) * W_ + x];
    tile[ty + j * 8][tx] = v;
  }
  __syncthreads();
#pragma unroll
  for (int j = 0; j < 4; ++j) {
    int x = x0 + ty + j * 8;
    int c = c0 + tx;
    if (x < W_)
      dst[(((size_t)b * H_ + y) * W_ + x) * C_ + c] = tile[tx][ty + j * 8];
  }
}

// ---------------------------------------------------------------------------
// Deformable RoI pooling. Block = (roi, bin); 256 threads = channels.
// Output layout [n, c*49 + bin] serves both xf[N,IN] and out[N,C,7,7].
// ---------------------------------------------------------------------------
__device__ __forceinline__ float pool_bin(const float* __restrict__ dataT,
                                          int b, int c, float wstart,
                                          float hstart, float sub_w,
                                          float sub_h) {
  float s = 0.f;
  int cnt = 0;
  const float* p = dataT + (size_t)b * H_ * W_ * C_;
#pragma unroll
  for (int sy = 0; sy < SAMPLE; ++sy) {
    float h = hstart + sy * sub_h;
#pragma unroll
    for (int sx = 0; sx < SAMPLE; ++sx) {
      float w = wstart + sx * sub_w;
      bool valid = (w >= -0.5f) && (w <= W_ - 0.5f) &&
                   (h >= -0.5f) && (h <= H_ - 0.5f);
      if (!valid) continue;  // uniform across block (coords don't depend on c)
      float wc = fminf(fmaxf(w, 0.f), (float)(W_ - 1));
      float hc = fminf(fmaxf(h, 0.f), (float)(H_ - 1));
      int x1 = (int)floorf(wc), x2 = (int)ceilf(wc);
      int y1 = (int)floorf(hc), y2 = (int)ceilf(hc);
      float dx = wc - (float)x1, dy = hc - (float)y1;
      float v11 = p[((size_t)y1 * W_ + x1) * C_ + c];
      float v12 = p[((size_t)y2 * W_ + x1) * C_ + c];
      float v21 = p[((size_t)y1 * W_ + x2) * C_ + c];
      float v22 = p[((size_t)y2 * W_ + x2) * C_ + c];
      s += (1.f - dx) * (1.f - dy) * v11 + (1.f - dx) * dy * v12 +
           dx * (1.f - dy) * v21 + dx * dy * v22;
      cnt++;
    }
  }
  return cnt > 0 ? s / (float)cnt : 0.f;
}

__global__ void __launch_bounds__(256)
deform_pool_kernel(const float* __restrict__ dataT,
                   const float* __restrict__ rois,
                   const float* __restrict__ trans,   // [N,2,7,7] or null
                   const float* __restrict__ maskv,   // [N,49] or null
                   float* __restrict__ out) {
  const int nb = blockIdx.x;
  const int n = nb / PP;
  const int bin = nb % PP;
  const int ph = bin / POOLED, pw = bin % POOLED;
  const int c = threadIdx.x;
  const float* r = rois + n * 5;
  const int b = (int)r[0];
  float roi_sw = rintf(r[1]) * SPATIAL_SCALE - 0.5f;
  float roi_sh = rintf(r[2]) * SPATIAL_SCALE - 0.5f;
  float roi_ew = (rintf(r[3]) + 1.f) * SPATIAL_SCALE - 0.5f;
  float roi_eh = (rintf(r[4]) + 1.f) * SPATIAL_SCALE - 0.5f;
  float roi_w = fmaxf(roi_ew - roi_sw, 0.1f);
  float roi_h = fmaxf(roi_eh - roi_sh, 0.1f);
  float bin_w = roi_w / POOLED, bin_h = roi_h / POOLED;
  float sub_w = bin_w / SAMPLE, sub_h = bin_h / SAMPLE;
  float tx = 0.f, ty = 0.f;
  if (trans) {
    tx = trans[(size_t)n * 2 * PP + bin] * TRANS_STD;
    ty = trans[(size_t)n * 2 * PP + PP + bin] * TRANS_STD;
  }
  float wstart = (float)pw * bin_w + roi_sw + tx * roi_w;
  float hstart = (float)ph * bin_h + roi_sh + ty * roi_h;
  float v = pool_bin(dataT, b, c, wstart, hstart, sub_w, sub_h);
  if (maskv) v *= maskv[(size_t)n * PP + bin];
  out[(size_t)n * IN + c * PP + bin] = v;
}

// ---------------------------------------------------------------------------
// GEMM: out[m,f] = act( sum_k A[m,k] * Wt[f,k] + bias[f] )
// A: [M=128, K] f32 row-major, Wt: [Nout, K] f32 row-major. K % 64 == 0.
// Block = 256 threads (8 waves) computing a 16(M) x 128(N) tile.
// Double-buffered LDS staging (f32 -> f16 convert in flight) feeding
// v_wmma_f32_16x16x32_f16 with f32 accumulation.
// act: 0=none, 1=relu, 2=sigmoid.
// ---------------------------------------------------------------------------
#define KT 64
#define LDSROW 72  // halves per LDS row (64 + pad; keeps 16B alignment)

__global__ void __launch_bounds__(256)
gemm_wmma_f16(const float* __restrict__ A, const float* __restrict__ Wt,
              const float* __restrict__ bias, float* __restrict__ out,
              int K, int Nout, int act) {
  __shared__ _Float16 Ah[2][16 * LDSROW];
  __shared__ _Float16 Bh[2][128 * LDSROW];
  const int tid = threadIdx.x;
  const int lane = tid & 31;
  const int wv = tid >> 5;            // wave 0..7 -> N sub-tile
  const int hl = lane >> 4;           // half-wave 0/1
  const int l16 = lane & 15;
  const int m0 = blockIdx.y * 16;
  const int fblock = blockIdx.x * 128;

  // --- staging coordinates (fixed per thread) ---
  // A tile: 16 x 64 halves; thread loads 4 consecutive floats of one row.
  const int arow = tid >> 4;                 // 0..15
  const int acol = (tid & 15) * 4;           // 0..60
  const float* asrc = A + (size_t)(m0 + arow) * K + acol;
  _Float16* adst0 = &Ah[0][arow * LDSROW + acol];
  _Float16* adst1 = &Ah[1][arow * LDSROW + acol];
  // B tile: 128 x 64 halves; thread loads 4 floats in each of 8 row-groups.
  // Branchless N-tail: clamp row (always in-bounds) and zero via keep factor.
  const int brow0 = tid >> 4;                // 0..15
  const int bcol = (tid & 15) * 4;
  const float* bsrc[8];
  float bkeep[8];
#pragma unroll
  for (int rb = 0; rb < 8; ++rb) {
    int f = fblock + rb * 16 + brow0;
    int fc = f < Nout ? f : (Nout - 1);
    bkeep[rb] = f < Nout ? 1.f : 0.f;
    bsrc[rb] = Wt + (size_t)fc * K + bcol;
  }

  float areg[4];
  float breg[8][4];
  v8f acc = {};

  const int nk = K / KT;

  // prologue: fetch + stage tile 0
#pragma unroll
  for (int j = 0; j < 4; ++j) areg[j] = asrc[j];
#pragma unroll
  for (int rb = 0; rb < 8; ++rb)
#pragma unroll
    for (int j = 0; j < 4; ++j) breg[rb][j] = bkeep[rb] * bsrc[rb][j];
#pragma unroll
  for (int j = 0; j < 4; ++j) adst0[j] = (_Float16)areg[j];
#pragma unroll
  for (int rb = 0; rb < 8; ++rb) {
    _Float16* d = &Bh[0][(rb * 16 + brow0) * LDSROW + bcol];
#pragma unroll
    for (int j = 0; j < 4; ++j) d[j] = (_Float16)breg[rb][j];
  }

  int cur = 0;
  for (int t = 0; t < nk; ++t) {
    __syncthreads();  // buffer `cur` ready; alternate buffer free

    // issue next tile's global loads while WMMAs run on buffer `cur`
    if (t + 1 < nk) {
      const int k0 = (t + 1) * KT;
#pragma unroll
      for (int j = 0; j < 4; ++j) areg[j] = asrc[k0 + j];
#pragma unroll
      for (int rb = 0; rb < 8; ++rb)
#pragma unroll
        for (int j = 0; j < 4; ++j) breg[rb][j] = bkeep[rb] * bsrc[rb][k0 + j];
    }

    // compute: 2 x v_wmma_f32_16x16x32_f16 over the 64-wide K window
    const _Float16* Ac = Ah[cur];
    const _Float16* Bc = Bh[cur];
#pragma unroll
    for (int kb = 0; kb < KT; kb += 32) {
      // A fragment (16x32 f16): lanes 0-15 hold K 0-7/16-23, lanes 16-31 hold
      // K 8-15/24-31 of the window (ISA 16-bit A layout).
      const _Float16* ap = Ac + l16 * LDSROW + kb + 8 * hl;
      v8h alo = *(const v8h*)(ap);
      v8h ahi = *(const v8h*)(ap + 16);
      // B fragment (32x16 f16): lanes 0-15 hold K 0-7/16-23, lanes 16-31 hold
      // K 8-15/24-31 (same k order as A).
      const _Float16* bp = Bc + (wv * 16 + l16) * LDSROW + kb + 8 * hl;
      v8h blo = *(const v8h*)(bp);
      v8h bhi = *(const v8h*)(bp + 16);
      v16h a, b;
#pragma unroll
      for (int i = 0; i < 8; ++i) {
        a[i] = alo[i]; a[8 + i] = ahi[i];
        b[i] = blo[i]; b[8 + i] = bhi[i];
      }
      acc = __builtin_amdgcn_wmma_f32_16x16x32_f16(
          false, a, false, b, (short)0, acc, false, false);
    }

    // stage next tile into the alternate buffer
    if (t + 1 < nk) {
      _Float16* ad = cur ? adst0 : adst1;
#pragma unroll
      for (int j = 0; j < 4; ++j) ad[j] = (_Float16)areg[j];
      _Float16* Bn = Bh[cur ^ 1];
#pragma unroll
      for (int rb = 0; rb < 8; ++rb) {
        _Float16* d = &Bn[(rb * 16 + brow0) * LDSROW + bcol];
#pragma unroll
        for (int j = 0; j < 4; ++j) d[j] = (_Float16)breg[rb][j];
      }
    }
    cur ^= 1;
  }

  // epilogue: C/D layout -> M = r + 8*hl, N = l16
#pragma unroll
  for (int r = 0; r < 8; ++r) {
    int m = m0 + r + 8 * hl;
    int f = fblock + wv * 16 + l16;
    if (f < Nout) {
      float v = acc[r] + bias[f];
      if (act == 1) v = fmaxf(v, 0.f);
      else if (act == 2) v = 1.f / (1.f + __expf(-v));
      out[(size_t)m * Nout + f] = v;
    }
  }
}

// ---------------------------------------------------------------------------
extern "C" void kernel_launch(void* const* d_in, const int* in_sizes, int n_in,
                              void* d_out, int out_size, void* d_ws,
                              size_t ws_size, hipStream_t stream) {
  const float* data = (const float*)d_in[0];
  const float* rois = (const float*)d_in[1];
  const float* w1  = (const float*)d_in[2];
  const float* b1  = (const float*)d_in[3];
  const float* w2  = (const float*)d_in[4];
  const float* b2  = (const float*)d_in[5];
  const float* w3  = (const float*)d_in[6];
  const float* b3  = (const float*)d_in[7];
  const float* mw1 = (const float*)d_in[8];
  const float* mb1 = (const float*)d_in[9];
  const float* mw2 = (const float*)d_in[10];
  const float* mb2 = (const float*)d_in[11];
  float* out = (float*)d_out;

  // workspace layout (floats): ~39.3 MB total
  float* ws = (float*)d_ws;
  size_t off = 0;
  float* dataT = ws + off; off += (size_t)B_ * H_ * W_ * C_;  // 7,782,400
  float* xf    = ws + off; off += (size_t)N_ * IN;            // 1,605,632
  float* h1    = ws + off; off += (size_t)N_ * FC;
  float* h2    = ws + off; off += (size_t)N_ * FC;
  float* m1    = ws + off; off += (size_t)N_ * FC;
  float* offs  = ws + off; off += (size_t)N_ * 2 * PP;
  float* maskv = ws + off; off += (size_t)N_ * PP;

  // 1. BCHW -> BHWC
  transpose_bchw_to_bhwc<<<dim3((W_ + 31) / 32, C_ / 32, B_ * H_),
                           dim3(32, 8), 0, stream>>>(data, dataT);
  // 2. first pooling (no trans) -> xf
  deform_pool_kernel<<<N_ * PP, C_, 0, stream>>>(dataT, rois, nullptr, nullptr,
                                                 xf);
  // 3-5. offset MLP
  gemm_wmma_f16<<<dim3(8, 8), 256, 0, stream>>>(xf, w1, b1, h1, IN, FC, 1);
  gemm_wmma_f16<<<dim3(8, 8), 256, 0, stream>>>(h1, w2, b2, h2, FC, FC, 1);
  gemm_wmma_f16<<<dim3(1, 8), 256, 0, stream>>>(h2, w3, b3, offs, FC, 2 * PP, 0);
  // 6-7. mask MLP
  gemm_wmma_f16<<<dim3(8, 8), 256, 0, stream>>>(xf, mw1, mb1, m1, IN, FC, 1);
  gemm_wmma_f16<<<dim3(1, 8), 256, 0, stream>>>(m1, mw2, mb2, maskv, FC, PP, 2);
  // 8. second pooling with offsets, scaled by mask -> out
  deform_pool_kernel<<<N_ * PP, C_, 0, stream>>>(dataT, rois, offs, maskv, out);
}
